// LoFTREncoderLayer_66202625900804
// MI455X (gfx1250) — matrix operand order repaired and checked by reading, not compiled, not run on hardware
//
#include <hip/hip_runtime.h>
#include <hip/hip_bf16.h>

// ---------------------------------------------------------------------------
// Types for CDNA5 WMMA (wave32): D(16x16,f32) = A(16x32,bf16) x B(32x16,bf16) + C
// ---------------------------------------------------------------------------
typedef __attribute__((ext_vector_type(16))) __bf16 v16bf;
typedef __attribute__((ext_vector_type(8)))  float  v8f;
typedef __attribute__((ext_vector_type(4)))  unsigned int u32x4;
typedef __attribute__((ext_vector_type(8)))  int i32x8;
typedef __attribute__((ext_vector_type(4)))  int i32x4;

__device__ __forceinline__ v8f wmma_bf16(v16bf a, v16bf b, v8f c) {
  return __builtin_amdgcn_wmma_f32_16x16x32_bf16(false, a, false, b, (short)0, c,
                                                 false, false);
}

// A-fragment (16x32 bf16, row-major, leading dim lda elements).
// ISA layout: lanes 0-15 -> M=0-15 (K=0..7,16..23), lanes 16-31 -> M=0-15 (K=8..15,24..31).
// Two contiguous 16B runs per lane -> two b128 loads.
__device__ __forceinline__ v16bf load_a16(const __bf16* base, int lda) {
  const int lane = threadIdx.x & 31;
  const int m = lane & 15, g = lane >> 4;
  const __bf16* p = base + (size_t)m * lda + 8 * g;
  union { uint4 u[2]; v16bf v; } t;
  t.u[0] = *(const uint4*)p;
  t.u[1] = *(const uint4*)(p + 16);
  return t.v;
}

// B-fragment (32x16 bf16, row-major [K][N], leading dim ldn).
// ISA layout: lanes 0-15 -> N=0-15 (K=0..7,16..23), lanes 16-31 -> N=0-15 (K=8..15,24..31).
__device__ __forceinline__ v16bf load_b16(const __bf16* base, int ldn) {
  const int lane = threadIdx.x & 31;
  const int n = lane & 15, g = lane >> 4;
  v16bf v;
#pragma unroll
  for (int j = 0; j < 16; ++j) {
    int k = ((j >> 3) << 4) + 8 * g + (j & 7);
    v[j] = base[(size_t)k * ldn + n];
  }
  return v;
}

// D layout: vgpr r, lane -> (m = r + 8*(lane>>4), n = lane&15)

// ---------------------------------------------------------------------------
// Problem constants
// ---------------------------------------------------------------------------
#define BN  4
#define BL  8192
#define BS  8192
#define BC  256
#define BH  512
#define NHEAD 8
#define HD  32
#define NL  (BN * BL)   // 32768
#define NS  (BN * BS)   // 32768

// ---------------------------------------------------------------------------
// TDM: DMA a ROWS x K bf16 tile (contiguous, lda==K) from global to LDS.
// Issued by one wave; caller must __syncthreads() after.
// ---------------------------------------------------------------------------
template <int K, int ROWS>
__device__ __forceinline__ void tdm_load_tile(const __bf16* gsrc, __bf16* lds_dst) {
  unsigned long long ga = (unsigned long long)(size_t)gsrc;
  unsigned int lds = (unsigned int)(size_t)lds_dst;
  u32x4 g0 = {1u, lds, (unsigned int)ga,
              (unsigned int)(((ga >> 32) & 0x1FFFFFFull) | (2u << 30))};
  i32x8 g1 = {(int)(1u << 16),                     // data_size=1 (2 bytes)
              (int)((unsigned)(K & 0xFFFF) << 16), // tensor_dim0 lo16
              0,                                   // dim0 hi16 | dim1 lo16
              (int)(16u | ((unsigned)K << 16)),    // dim1 hi16 (1<<20) | tile_dim0=K
              ROWS,                                // tile_dim1=ROWS | tile_dim2=0
              K,                                   // tensor_dim0_stride lo32
              0, 0};
  i32x4 gz4 = {0, 0, 0, 0};
  i32x8 gz8 = {0, 0, 0, 0, 0, 0, 0, 0};
  __builtin_amdgcn_tensor_load_to_lds(g0, g1, gz4, gz4, gz8, 0);
  __builtin_amdgcn_s_wait_tensorcnt(0);
}

// ---------------------------------------------------------------------------
// Small utility kernels
// ---------------------------------------------------------------------------
__global__ void __launch_bounds__(256) wt_transpose_kernel(
    const float* __restrict__ W, __bf16* __restrict__ Wt, int No, int Ni) {
  int idx = blockIdx.x * 256 + threadIdx.x;
  if (idx >= No * Ni) return;
  int co = idx / Ni, ci = idx - co * Ni;
  Wt[(size_t)ci * No + co] = (__bf16)W[idx];
}

__global__ void __launch_bounds__(256) cvt_bf16_kernel(
    const float* __restrict__ src, __bf16* __restrict__ dst, int n) {
  int idx = blockIdx.x * 256 + threadIdx.x;
  if (idx < n) dst[idx] = (__bf16)src[idx];
}

__global__ void __launch_bounds__(256) zero_f32_kernel(float* __restrict__ p, int n) {
  int idx = blockIdx.x * 256 + threadIdx.x;
  if (idx < n) p[idx] = 0.f;
}

// Ksum[n][c] = sum_s Kf[n][s][c]
__global__ void __launch_bounds__(256) ksum_kernel(
    const __bf16* __restrict__ Kf, float* __restrict__ Ksum) {
  int idx = blockIdx.x * 256 + threadIdx.x;   // N*C = 1024 threads
  int n = idx >> 8, c = idx & 255;
  const __bf16* p = Kf + (size_t)n * BS * BC + c;
  float s = 0.f;
  for (int ss = 0; ss < BS; ++ss) s += (float)p[(size_t)ss * BC];
  Ksum[idx] = s;
}

// Z[n][l][h] = 1 / (dot(Qf[n,l,h*32:+32], Ksum[n,h*32:+32]) + 1e-6)
__global__ void __launch_bounds__(256) z_kernel(
    const __bf16* __restrict__ Qf, const float* __restrict__ Ksum,
    float* __restrict__ Z) {
  int idx = blockIdx.x * 256 + threadIdx.x;   // NL * NHEAD
  int row = idx >> 3, h = idx & 7;
  int n = row >> 13;                          // row / BL
  const __bf16* q = Qf + (size_t)row * BC + h * HD;
  const float* ks = Ksum + n * BC + h * HD;
  float dot = 0.f;
#pragma unroll
  for (int j = 0; j < HD; ++j) dot += (float)q[j] * ks[j];
  Z[idx] = 1.f / (dot + 1e-6f);
}

// ---------------------------------------------------------------------------
// GEMM: out[M x NOUT] = epi(A[M x K] * BT[K x NOUT])   (bf16 in, f32 accum)
// 256 threads = 8 waves; block computes 32 rows x NOUT; each wave: 2 row-tiles
// x TPW col-tiles, so each A/B fragment feeds multiple WMMAs.
// EPI: 0 = elu(x)+1,  1 = x*scale,  2 = none
// ---------------------------------------------------------------------------
template <int K, int NOUT, int EPI>
__global__ void __launch_bounds__(256) gemm_epi_kernel(
    const __bf16* __restrict__ A, const __bf16* __restrict__ BT,
    __bf16* __restrict__ out, float scale) {
  __shared__ __bf16 sA[32 * K];
  const int m0 = blockIdx.x * 32;
  if (threadIdx.x < 32) {           // wave 0 drives the Tensor Data Mover
    tdm_load_tile<K, 32>(A + (size_t)m0 * K, sA);
  }
  __syncthreads();

  const int w = threadIdx.x >> 5;
  constexpr int TPW = NOUT / 128;            // 16-col tiles per wave
  v8f acc[2][TPW] = {};
#pragma unroll 1
  for (int kc = 0; kc < K / 32; ++kc) {
    v16bf a0 = load_a16(sA + kc * 32, K);
    v16bf a1 = load_a16(sA + 16 * K + kc * 32, K);
    if (kc + 1 < K / 32)
      __builtin_prefetch(BT + (size_t)((kc + 1) * 32) * NOUT, 0, 3);
#pragma unroll
    for (int t = 0; t < TPW; ++t) {
      int n0 = (w * TPW + t) * 16;
      v16bf b = load_b16(BT + (size_t)(kc * 32) * NOUT + n0, NOUT);
      acc[0][t] = wmma_bf16(a0, b, acc[0][t]);
      acc[1][t] = wmma_bf16(a1, b, acc[1][t]);
    }
  }
  const int lane = threadIdx.x & 31, g = lane >> 4, ln = lane & 15;
#pragma unroll
  for (int rt = 0; rt < 2; ++rt) {
#pragma unroll
    for (int t = 0; t < TPW; ++t) {
      int n0 = (w * TPW + t) * 16;
#pragma unroll
      for (int r = 0; r < 8; ++r) {
        float v = acc[rt][t][r];
        if (EPI == 0) v = (v > 0.f) ? (v + 1.f) : __expf(v);    // elu(x)+1
        else if (EPI == 1) v = v * scale;
        out[(size_t)(m0 + rt * 16 + r + 8 * g) * NOUT + n0 + ln] = (__bf16)v;
      }
    }
  }
}

// ---------------------------------------------------------------------------
// KV[n][h][d][v] = sum_s Kf[n,s,h*32+d] * Vs[n,s,h*32+v]  (split-K + atomics)
// ---------------------------------------------------------------------------
__global__ void __launch_bounds__(256) kv_kernel(
    const __bf16* __restrict__ Kf, const __bf16* __restrict__ Vs,
    float* __restrict__ KV) {
  const int gw = blockIdx.x * 8 + (threadIdx.x >> 5);   // 0..4095
  const int sp = gw & 127;                              // 128 splits
  const int pair = gw >> 7;                             // n*8+h, 0..31
  const int h = pair & 7, n = pair >> 3;
  const int s0 = sp * 64;
  const int lane = threadIdx.x & 31, g = lane >> 4, ln = lane & 15;
  const int c0 = h * HD;
  const size_t base = (size_t)n * BS * BC;

  v8f acc[2][2] = {};
#pragma unroll
  for (int kc = 0; kc < 2; ++kc) {
    int s = s0 + kc * 32;
    v16bf a[2];
#pragma unroll
    for (int mt = 0; mt < 2; ++mt) {
#pragma unroll
      for (int j = 0; j < 16; ++j) {
        int k = ((j >> 3) << 4) + 8 * g + (j & 7);      // ISA A-layout k index
        a[mt][j] = Kf[base + (size_t)(s + k) * BC + c0 + mt * 16 + ln];
      }
    }
#pragma unroll
    for (int vt = 0; vt < 2; ++vt) {
      v16bf b = load_b16(Vs + base + (size_t)s * BC + c0 + vt * 16, BC);
      acc[0][vt] = wmma_bf16(a[0], b, acc[0][vt]);
      acc[1][vt] = wmma_bf16(a[1], b, acc[1][vt]);
    }
  }
  float* kvp = KV + (size_t)pair * (HD * HD);
#pragma unroll
  for (int mt = 0; mt < 2; ++mt)
#pragma unroll
    for (int vt = 0; vt < 2; ++vt)
#pragma unroll
      for (int r = 0; r < 8; ++r)
        atomicAdd(&kvp[(mt * 16 + r + 8 * g) * HD + vt * 16 + ln], acc[mt][vt][r]);
}

// ---------------------------------------------------------------------------
// message[n,l,h,v] = (Qf[n,l,h,:] @ KVb[n,h,:,v]) * Z[n,l,h] * S
// ---------------------------------------------------------------------------
__global__ void __launch_bounds__(256) message_kernel(
    const __bf16* __restrict__ Qf, const __bf16* __restrict__ KVb,
    const float* __restrict__ Z, __bf16* __restrict__ msg) {
  const int gw = blockIdx.x * 8 + (threadIdx.x >> 5);
  const int LT = BL / 16;                               // 512
  const int lt = gw % LT;
  const int rest = gw / LT;
  const int h = rest & 7, n = rest >> 3;
  const int row0 = n * BL + lt * 16;

  v16bf a = load_a16(Qf + (size_t)row0 * BC + h * HD, BC);
  const __bf16* kvb = KVb + (size_t)(n * NHEAD + h) * (HD * HD);
  v8f acc0 = {}, acc1 = {};
  acc0 = wmma_bf16(a, load_b16(kvb, HD), acc0);
  acc1 = wmma_bf16(a, load_b16(kvb + 16, HD), acc1);

  const int lane = threadIdx.x & 31, g = lane >> 4, ln = lane & 15;
#pragma unroll
  for (int r = 0; r < 8; ++r) {
    int m = r + 8 * g;
    float z = Z[(size_t)(row0 + m) * NHEAD + h] * (float)BS;
    size_t o = (size_t)(row0 + m) * BC + h * HD;
    msg[o + ln] = (__bf16)(acc0[r] * z);
    msg[o + 16 + ln] = (__bf16)(acc1[r] * z);
  }
}

// ---------------------------------------------------------------------------
// merge+LN: mln = LayerNorm(msg @ WmT, ln1) as bf16.  Block = 32 rows x 256.
// ---------------------------------------------------------------------------
__global__ void __launch_bounds__(256) merge_ln_kernel(
    const __bf16* __restrict__ msg, const __bf16* __restrict__ WmT,
    const float* __restrict__ gamma, const float* __restrict__ beta,
    __bf16* __restrict__ mln) {
  __shared__ __bf16 sA[32 * BC];
  __shared__ float sC[32 * BC];
  __shared__ float red[512];
  const int m0 = blockIdx.x * 32;
  if (threadIdx.x < 32) {
    tdm_load_tile<BC, 32>(msg + (size_t)m0 * BC, sA);
  }
  __syncthreads();

  const int w = threadIdx.x >> 5, lane = threadIdx.x & 31;
  const int g = lane >> 4, ln = lane & 15;
  v8f acc[2][2] = {};
#pragma unroll 1
  for (int kc = 0; kc < BC / 32; ++kc) {
    v16bf a0 = load_a16(sA + kc * 32, BC);
    v16bf a1 = load_a16(sA + 16 * BC + kc * 32, BC);
#pragma unroll
    for (int t = 0; t < 2; ++t) {
      int n0 = (w * 2 + t) * 16;
      v16bf b = load_b16(WmT + (size_t)(kc * 32) * BC + n0, BC);
      acc[0][t] = wmma_bf16(a0, b, acc[0][t]);
      acc[1][t] = wmma_bf16(a1, b, acc[1][t]);
    }
  }
#pragma unroll
  for (int rt = 0; rt < 2; ++rt)
#pragma unroll
    for (int t = 0; t < 2; ++t) {
      int n0 = (w * 2 + t) * 16;
#pragma unroll
      for (int r = 0; r < 8; ++r)
        sC[(rt * 16 + r + 8 * g) * BC + n0 + ln] = acc[rt][t][r];
    }
  __syncthreads();

  // LayerNorm over each row of 256: 8 threads x 32 cols per row
  const int row = threadIdx.x >> 3, sub = threadIdx.x & 7;
  float s = 0.f, ss = 0.f;
#pragma unroll
  for (int j = 0; j < 32; ++j) {
    float v = sC[row * BC + sub * 32 + j];
    s += v; ss += v * v;
  }
  red[row * 8 + sub] = s;
  red[256 + row * 8 + sub] = ss;
  __syncthreads();
  if (sub == 0) {
    float S1 = 0.f, S2 = 0.f;
#pragma unroll
    for (int j = 0; j < 8; ++j) { S1 += red[row * 8 + j]; S2 += red[256 + row * 8 + j]; }
    float mu = S1 * (1.f / BC);
    float var = S2 * (1.f / BC) - mu * mu;
    red[row * 8] = mu;
    red[256 + row * 8] = rsqrtf(var + 1e-5f);
  }
  __syncthreads();
  float mu = red[row * 8], rs = red[256 + row * 8];
#pragma unroll
  for (int j = 0; j < 32; ++j) {
    int c = sub * 32 + j;
    float v = (sC[row * BC + c] - mu) * rs * gamma[c] + beta[c];
    mln[(size_t)(m0 + row) * BC + c] = (__bf16)v;
  }
}

// ---------------------------------------------------------------------------
// Fused gated MLP: h = relu([xb, mln] @ W1T); y = LN2(h @ W2T); out = x + y
// Block = 32 rows; concat input and h live in LDS.
// ---------------------------------------------------------------------------
__global__ void __launch_bounds__(256) mlp_kernel(
    const __bf16* __restrict__ xb, const __bf16* __restrict__ mln,
    const __bf16* __restrict__ W1T, const __bf16* __restrict__ W2T,
    const float* __restrict__ gamma, const float* __restrict__ beta,
    const float* __restrict__ x, float* __restrict__ out) {
  __shared__ __bf16 sA[32 * BH];
  __shared__ float sC[32 * BC];
  __shared__ float red[512];
  const int m0 = blockIdx.x * 32;
  {
    const uint4* gx = (const uint4*)(xb + (size_t)m0 * BC);
    const uint4* gm = (const uint4*)(mln + (size_t)m0 * BC);
    for (int c = threadIdx.x; c < 1024; c += 256) {
      int row = c >> 5, rc = c & 31;
      ((uint4*)(sA + row * BH))[rc] = gx[row * 32 + rc];
      ((uint4*)(sA + row * BH + BC))[rc] = gm[row * 32 + rc];
    }
  }
  __syncthreads();

  const int w = threadIdx.x >> 5, lane = threadIdx.x & 31;
  const int g = lane >> 4, ln = lane & 15;

  // GEMM1: [32 x 512] x [512 x 512]
  v8f acc1[2][4] = {};
#pragma unroll 1
  for (int kc = 0; kc < BH / 32; ++kc) {
    v16bf a0 = load_a16(sA + kc * 32, BH);
    v16bf a1 = load_a16(sA + 16 * BH + kc * 32, BH);
    if (kc + 1 < BH / 32)
      __builtin_prefetch(W1T + (size_t)((kc + 1) * 32) * BH, 0, 3);
#pragma unroll
    for (int t = 0; t < 4; ++t) {
      int n0 = (w * 4 + t) * 16;
      v16bf b = load_b16(W1T + (size_t)(kc * 32) * BH + n0, BH);
      acc1[0][t] = wmma_bf16(a0, b, acc1[0][t]);
      acc1[1][t] = wmma_bf16(a1, b, acc1[1][t]);
    }
  }
  __syncthreads();
  // ReLU -> h (bf16) back into sA
#pragma unroll
  for (int rt = 0; rt < 2; ++rt)
#pragma unroll
    for (int t = 0; t < 4; ++t) {
      int n0 = (w * 4 + t) * 16;
#pragma unroll
      for (int r = 0; r < 8; ++r)
        sA[(rt * 16 + r + 8 * g) * BH + n0 + ln] = (__bf16)fmaxf(acc1[rt][t][r], 0.f);
    }
  __syncthreads();

  // GEMM2: [32 x 512] x [512 x 256]
  v8f acc2[2][2] = {};
#pragma unroll 1
  for (int kc = 0; kc < BH / 32; ++kc) {
    v16bf a0 = load_a16(sA + kc * 32, BH);
    v16bf a1 = load_a16(sA + 16 * BH + kc * 32, BH);
#pragma unroll
    for (int t = 0; t < 2; ++t) {
      int n0 = (w * 2 + t) * 16;
      v16bf b = load_b16(W2T + (size_t)(kc * 32) * BC + n0, BC);
      acc2[0][t] = wmma_bf16(a0, b, acc2[0][t]);
      acc2[1][t] = wmma_bf16(a1, b, acc2[1][t]);
    }
  }
#pragma unroll
  for (int rt = 0; rt < 2; ++rt)
#pragma unroll
    for (int t = 0; t < 2; ++t) {
      int n0 = (w * 2 + t) * 16;
#pragma unroll
      for (int r = 0; r < 8; ++r)
        sC[(rt * 16 + r + 8 * g) * BC + n0 + ln] = acc2[rt][t][r];
    }
  __syncthreads();

  // LayerNorm + residual: 8 threads x 32 cols per row
  const int row = threadIdx.x >> 3, sub = threadIdx.x & 7;
  float s = 0.f, ss = 0.f;
#pragma unroll
  for (int j = 0; j < 32; ++j) {
    float v = sC[row * BC + sub * 32 + j];
    s += v; ss += v * v;
  }
  red[row * 8 + sub] = s;
  red[256 + row * 8 + sub] = ss;
  __syncthreads();
  if (sub == 0) {
    float S1 = 0.f, S2 = 0.f;
#pragma unroll
    for (int j = 0; j < 8; ++j) { S1 += red[row * 8 + j]; S2 += red[256 + row * 8 + j]; }
    float mu = S1 * (1.f / BC);
    float var = S2 * (1.f / BC) - mu * mu;
    red[row * 8] = mu;
    red[256 + row * 8] = rsqrtf(var + 1e-5f);
  }
  __syncthreads();
  float mu = red[row * 8], rs = red[256 + row * 8];
#pragma unroll
  for (int j = 0; j < 32; ++j) {
    int c = sub * 32 + j;
    float v = (sC[row * BC + c] - mu) * rs * gamma[c] + beta[c];
    size_t o = (size_t)(m0 + row) * BC + c;
    out[o] = x[o] + v;
  }
}

// ---------------------------------------------------------------------------
// Host launch
// ---------------------------------------------------------------------------
extern "C" void kernel_launch(void* const* d_in, const int* in_sizes, int n_in,
                              void* d_out, int out_size, void* d_ws, size_t ws_size,
                              hipStream_t stream) {
  const float* x    = (const float*)d_in[0];
  const float* src  = (const float*)d_in[1];
  const float* Wq   = (const float*)d_in[2];
  const float* Wk   = (const float*)d_in[3];
  const float* Wv   = (const float*)d_in[4];
  const float* Wm   = (const float*)d_in[5];
  const float* g1   = (const float*)d_in[6];
  const float* b1   = (const float*)d_in[7];
  const float* W1   = (const float*)d_in[8];
  const float* W2   = (const float*)d_in[9];
  const float* g2   = (const float*)d_in[10];
  const float* b2   = (const float*)d_in[11];
  float* out = (float*)d_out;

  constexpr size_t CC2  = (size_t)BC * BC * 2;
  constexpr size_t HH2  = (size_t)BH * BH * 2;
  constexpr size_t HC2  = (size_t)BH * BC * 2;
  constexpr size_t NLC2 = (size_t)NL * BC * 2;

  char* ws = (char*)d_ws;
  size_t o = 0;
  __bf16* WqT = (__bf16*)(ws + o); o += CC2;
  __bf16* WkT = (__bf16*)(ws + o); o += CC2;
  __bf16* WvT = (__bf16*)(ws + o); o += CC2;
  __bf16* WmT = (__bf16*)(ws + o); o += CC2;
  __bf16* W1T = (__bf16*)(ws + o); o += HH2;
  __bf16* W2T = (__bf16*)(ws + o); o += HC2;
  __bf16* XB  = (__bf16*)(ws + o); o += NLC2;
  __bf16* SB  = (__bf16*)(ws + o); o += NLC2;   // reused later as MSG
  __bf16* QF  = (__bf16*)(ws + o); o += NLC2;
  __bf16* KF  = (__bf16*)(ws + o); o += NLC2;   // reused later as MLN
  __bf16* VS  = (__bf16*)(ws + o); o += NLC2;
  float*  KSUM = (float*)(ws + o); o += (size_t)BN * BC * 4;
  float*  KV   = (float*)(ws + o); o += (size_t)BN * NHEAD * HD * HD * 4;
  __bf16* KVB  = (__bf16*)(ws + o); o += (size_t)BN * NHEAD * HD * HD * 2;
  float*  ZB   = (float*)(ws + o); o += (size_t)NL * NHEAD * 4;
  __bf16* MSG = SB;
  __bf16* MLN = KF;

  wt_transpose_kernel<<<(BC * BC) / 256, 256, 0, stream>>>(Wq, WqT, BC, BC);
  wt_transpose_kernel<<<(BC * BC) / 256, 256, 0, stream>>>(Wk, WkT, BC, BC);
  wt_transpose_kernel<<<(BC * BC) / 256, 256, 0, stream>>>(Wv, WvT, BC, BC);
  wt_transpose_kernel<<<(BC * BC) / 256, 256, 0, stream>>>(Wm, WmT, BC, BC);
  wt_transpose_kernel<<<(BH * BH) / 256, 256, 0, stream>>>(W1, W1T, BH, BH);
  wt_transpose_kernel<<<(BC * BH) / 256, 256, 0, stream>>>(W2, W2T, BC, BH);

  cvt_bf16_kernel<<<(NL * BC) / 256, 256, 0, stream>>>(x, XB, NL * BC);
  cvt_bf16_kernel<<<(NS * BC) / 256, 256, 0, stream>>>(src, SB, NS * BC);

  gemm_epi_kernel<BC, BC, 0><<<NL / 32, 256, 0, stream>>>(XB, WqT, QF, 0.f);
  gemm_epi_kernel<BC, BC, 0><<<NS / 32, 256, 0, stream>>>(SB, WkT, KF, 0.f);
  gemm_epi_kernel<BC, BC, 1><<<NS / 32, 256, 0, stream>>>(SB, WvT, VS, 1.f / (float)BS);

  zero_f32_kernel<<<(BN * NHEAD * HD * HD) / 256, 256, 0, stream>>>(
      KV, BN * NHEAD * HD * HD);
  ksum_kernel<<<(BN * BC) / 256, 256, 0, stream>>>(KF, KSUM);
  kv_kernel<<<512, 256, 0, stream>>>(KF, VS, KV);
  cvt_bf16_kernel<<<(BN * NHEAD * HD * HD) / 256, 256, 0, stream>>>(
      KV, KVB, BN * NHEAD * HD * HD);
  z_kernel<<<(NL * NHEAD) / 256, 256, 0, stream>>>(QF, KSUM, ZB);
  message_kernel<<<(BN * NHEAD * (BL / 16)) / 8, 256, 0, stream>>>(QF, KVB, ZB, MSG);

  merge_ln_kernel<<<NL / 32, 256, 0, stream>>>(MSG, WmT, g1, b1, MLN);

  mlp_kernel<<<NL / 32, 256, 0, stream>>>(XB, MLN, W1T, W2T, g2, b2, x, out);
}
